// FeatureExtractor_11587821765039
// MI455X (gfx1250) — matrix operand order repaired and checked by reading, not compiled, not run on hardware
//
#include <hip/hip_runtime.h>
#include <hip/hip_bf16.h>

// ---------------- types for WMMA ----------------
typedef __attribute__((ext_vector_type(16))) __bf16 v16bf;
typedef __attribute__((ext_vector_type(8)))  float  v8f;
typedef unsigned short bh;   // raw bf16 bits in memory

union Frag { v16bf v; unsigned int u[8]; };

__device__ __forceinline__ bh f2bf(float f) {
    unsigned u = __float_as_uint(f);
    unsigned r = u + 0x7FFFu + ((u >> 16) & 1u);   // round-to-nearest-even
    return (bh)(r >> 16);
}

// sizes
#define BB   16
#define TT   256
#define LL   1024
#define NF   32
#define HH   1024
#define GIN  96
#define BT   (BB*TT)       // 4096
#define G3   (3*HH)        // 3072

// persistent GRU kernel geometry
#define GRU_BLOCKS   24
#define GRU_THREADS  256
#define GRU_NT       (GRU_BLOCKS * GRU_THREADS)   // 6144 threads, 192 waves
#define SROW         (HH + 8)                     // padded LDS row stride (bh elems)

// ---------------- fragment loaders (ISA 7.12.2 layouts) ----------------
__device__ __forceinline__ void load_afrag(Frag& a, const bh* __restrict__ A,
                                           int lda, int m0, int k0, int r16, int half) {
    const bh* ap = A + (size_t)(m0 + r16) * lda + k0 + half * 8;
#pragma unroll
    for (int v = 0; v < 4; ++v) {
        a.u[v]     = *(const unsigned int*)(ap + 2 * v);
        a.u[v + 4] = *(const unsigned int*)(ap + 16 + 2 * v);
    }
}
__device__ __forceinline__ void load_bfrag(Frag& b, const bh* __restrict__ W,
                                           int ldw, int n0, int k0, int r16, int half) {
    const bh* wp = W + (size_t)(n0 + r16) * ldw + k0 + half * 8;
#pragma unroll
    for (int v = 0; v < 4; ++v) {
        b.u[v]     = *(const unsigned int*)(wp + 2 * v);
        b.u[v + 4] = *(const unsigned int*)(wp + 16 + 2 * v);
    }
}
// A-fragment from LDS (padded rows): emits ds_load_b128 pairs
__device__ __forceinline__ void load_afrag_lds(Frag& a, const bh* sh,
                                               int k0, int r16, int half) {
    const bh* ap = sh + r16 * SROW + k0 + half * 8;
#pragma unroll
    for (int v = 0; v < 4; ++v) {
        a.u[v]     = *(const unsigned int*)(ap + 2 * v);
        a.u[v + 4] = *(const unsigned int*)(ap + 16 + 2 * v);
    }
}

// ---------------- small utility kernels ----------------
__global__ void k_cvt_bf16(const float* __restrict__ src, bh* __restrict__ dst, int n) {
    int i = blockIdx.x * blockDim.x + threadIdx.x;
    if (i < n) dst[i] = f2bf(src[i]);
}

// dst[c*R + r] = bf16(src[r*C + c])  (transpose while converting)
__global__ void k_cvt_tr_bf16(const float* __restrict__ src, bh* __restrict__ dst, int R, int C) {
    int i = blockIdx.x * blockDim.x + threadIdx.x;
    if (i < R * C) {
        int r = i / C, c = i % C;
        dst[(size_t)c * R + r] = f2bf(src[i]);
    }
}

__global__ void k_zero_bar(int* __restrict__ bar) { bar[0] = 0; bar[1] = 0; }

__global__ void k_ones(float* __restrict__ p, int n) {
    int i = blockIdx.x * blockDim.x + threadIdx.x;
    if (i < n) p[i] = 1.0f;
}

// ---------------- conv branches + relu + global max pool ----------------
__global__ void k_conv_pool(const float* __restrict__ x,
                            const float* __restrict__ w3, const float* __restrict__ b3,
                            const float* __restrict__ w5, const float* __restrict__ b5,
                            const float* __restrict__ w7, const float* __restrict__ b7,
                            bh* __restrict__ featB) {
    __shared__ float sx[LL];
    const float* xr = x + (size_t)blockIdx.x * LL;
    for (int i = threadIdx.x; i < LL; i += blockDim.x) sx[i] = xr[i];
    __syncthreads();

    int f = threadIdx.x;
    if (f < GIN) {
        int g = f / NF, ff = f % NF;
        const float* w; const float* bb; int K;
        if (g == 0)      { w = w3 + ff * 3; bb = b3; K = 3; }
        else if (g == 1) { w = w5 + ff * 5; bb = b5; K = 5; }
        else             { w = w7 + ff * 7; bb = b7; K = 7; }
        float wr[7];
        for (int k = 0; k < K; ++k) wr[k] = w[k];
        float bias = bb[ff];
        int pad = K >> 1;
        float m = 0.0f;
        for (int p = 0; p < LL; ++p) {
            float s = bias;
            int lo = p - pad;
            for (int k = 0; k < K; ++k) {
                int idx = lo + k;
                if (idx >= 0 && idx < LL) s += sx[idx] * wr[k];
            }
            m = fmaxf(m, fmaxf(s, 0.0f));
        }
        featB[(size_t)blockIdx.x * GIN + f] = f2bf(m);
    }
}

// ---------------- wide bf16 WMMA GEMM:  C[MxN] = A[MxK] * W[NxK]^T  ----------------
// blockDim = 128 (4 waves). Each wave computes a 16x64 strip.
template<int EPI>
__global__ void k_gemm_bf16(const bh* __restrict__ A, int lda,
                            const bh* __restrict__ W, int ldw,
                            const float* __restrict__ bias,
                            const float* __restrict__ res,
                            float* __restrict__ Cf, bh* __restrict__ Cb,
                            int ldc, int K) {
    const int wave = threadIdx.x >> 5;
    const int lane = threadIdx.x & 31;
    const int half = lane >> 4;
    const int r16  = lane & 15;
    const int n0   = (blockIdx.x * 4 + wave) * 64;
    const int m0   = blockIdx.y * 16;

    v8f acc0 = {}, acc1 = {}, acc2 = {}, acc3 = {};
    for (int k0 = 0; k0 < K; k0 += 32) {
        Frag a, b0, b1, b2, b3;
        load_afrag(a, A, lda, m0, k0, r16, half);
        load_bfrag(b0, W, ldw, n0,      k0, r16, half);
        load_bfrag(b1, W, ldw, n0 + 16, k0, r16, half);
        load_bfrag(b2, W, ldw, n0 + 32, k0, r16, half);
        load_bfrag(b3, W, ldw, n0 + 48, k0, r16, half);
        acc0 = __builtin_amdgcn_wmma_f32_16x16x32_bf16(false, a.v, false, b0.v, (short)0, acc0, false, false);
        acc1 = __builtin_amdgcn_wmma_f32_16x16x32_bf16(false, a.v, false, b1.v, (short)0, acc1, false, false);
        acc2 = __builtin_amdgcn_wmma_f32_16x16x32_bf16(false, a.v, false, b2.v, (short)0, acc2, false, false);
        acc3 = __builtin_amdgcn_wmma_f32_16x16x32_bf16(false, a.v, false, b3.v, (short)0, acc3, false, false);
    }

    v8f accs[4] = {acc0, acc1, acc2, acc3};
#pragma unroll
    for (int j = 0; j < 4; ++j) {
        const int n = n0 + j * 16 + r16;
#pragma unroll
        for (int r = 0; r < 8; ++r) {
            const int m = m0 + half * 8 + r;
            const size_t o = (size_t)m * ldc + n;
            const float v = accs[j][r];
            if (EPI == 0) {
                Cf[o] = v + bias[n];
            } else if (EPI == 1) {
                Cb[o] = f2bf(res[o] - fmaxf(v, 0.0f));
            } else {
                Cb[o] = f2bf(fmaxf(v + bias[n], 0.0f));
            }
        }
    }
}

// ---------------- software grid barrier (agent scope) ----------------
__device__ __forceinline__ void grid_barrier(int* __restrict__ bar, int nblk) {
    __threadfence();
    __syncthreads();
    if (threadIdx.x == 0) {
        int* cnt = bar;
        int* gen = bar + 1;
        int g = __hip_atomic_load(gen, __ATOMIC_ACQUIRE, __HIP_MEMORY_SCOPE_AGENT);
        int a = __hip_atomic_fetch_add(cnt, 1, __ATOMIC_ACQ_REL, __HIP_MEMORY_SCOPE_AGENT);
        if (a == nblk - 1) {
            __hip_atomic_store(cnt, 0, __ATOMIC_RELAXED, __HIP_MEMORY_SCOPE_AGENT);
            __hip_atomic_store(gen, g + 1, __ATOMIC_RELEASE, __HIP_MEMORY_SCOPE_AGENT);
        } else {
            while (__hip_atomic_load(gen, __ATOMIC_ACQUIRE, __HIP_MEMORY_SCOPE_AGENT) == g)
                __builtin_amdgcn_s_sleep(1);
        }
    }
    __syncthreads();
}

// ---------------- persistent GRU: all 256 timesteps in ONE launch ----------------
// 24 blocks x 256 threads = 192 waves. W_hh B-fragments are loaded ONCE into a
// 256-VGPR register file slice (Frag bfr[32], fully-unrolled indices) and reused
// across all 256 timesteps. Per step:
//   stage:   block DMA-copies h (32 KB) global->LDS via global_load_async_to_lds_b128
//   phase 1: 192 waves each compute one 16x16 tile of gh = h @ W_hh^T + b_hh
//            (A software-pipelined from padded LDS through TWO distinct named
//             buffers -> partial s_wait_dscnt; B from the persistent registers)
//   phase 2: 6144 threads fuse the gates; h carry kept in registers per thread
__global__ void __launch_bounds__(GRU_THREADS)
k_gru_persist(const bh* __restrict__ WhhB, const float* __restrict__ bhh,
              const float* __restrict__ gatesX,
              bh* __restrict__ hB, float* __restrict__ hsF, bh* __restrict__ hsB,
              float* __restrict__ gh, int* __restrict__ bar) {
    __shared__ bh sh[BB * SROW];                 // 33 KB staged h, padded rows

    const int gtid  = blockIdx.x * GRU_THREADS + threadIdx.x;
    const int gwave = gtid >> 5;                 // 0..191
    const int lane  = threadIdx.x & 31;
    const int half  = lane >> 4;
    const int r16   = lane & 15;
    const int n0    = gwave * 16;                // this wave's gh tile column

    const unsigned ldsbase = (unsigned)(size_t)&sh[0];
    const unsigned long long gbase = (unsigned long long)hB;

    // ---- load this wave's entire W_hh slice into registers (once) ----
    Frag bfr[32];
#pragma unroll
    for (int kk = 0; kk < 32; ++kk)
        load_bfrag(bfr[kk], WhhB, HH, n0, kk * 32, r16, half);

    // zero the h carry (registers + hB matrix), then sync the grid once
    float hreg[3] = {0.0f, 0.0f, 0.0f};
    for (int idx = gtid; idx < BB * HH; idx += GRU_NT) hB[idx] = (bh)0;
    grid_barrier(bar, GRU_BLOCKS);

    for (int t = 0; t < TT; ++t) {
        // ---- stage h into LDS: 2048 x b128 async DMA chunks per block ----
#pragma unroll
        for (int it = 0; it < 8; ++it) {
            int c   = threadIdx.x + it * GRU_THREADS;          // 0..2047
            int row = c >> 7;                                  // 16 rows
            int col = c & 127;                                 // 128 x 16B per row
            unsigned lds = ldsbase + (unsigned)(row * (SROW * 2) + col * 16);
            unsigned gof = (unsigned)(row * (HH * 2) + col * 16);
            asm volatile("global_load_async_to_lds_b128 %0, %1, %2"
                         :: "v"(lds), "v"(gof), "s"(gbase) : "memory");
        }
        asm volatile("s_wait_asynccnt 0x0" ::: "memory");
        __syncthreads();

        // ---- phase 1: gh(16x3072) = h(16x1024) @ WhhB^T + bhh ----
        // A software-pipelined through two DISTINCT named buffers (a0/a1) so the
        // register allocator cannot merge them; the next fragment's ds_loads are
        // issued before the current WMMA -> partial dscnt waits, latency hidden.
        v8f acc = {};
        Frag a0, a1;
        load_afrag_lds(a0, sh, 0, r16, half);
#pragma unroll
        for (int kk = 0; kk < 32; kk += 2) {
            load_afrag_lds(a1, sh, (kk + 1) * 32, r16, half);
            acc = __builtin_amdgcn_wmma_f32_16x16x32_bf16(false, a0.v, false,
                                                          bfr[kk].v, (short)0, acc,
                                                          false, false);
            if (kk + 2 < 32)
                load_afrag_lds(a0, sh, (kk + 2) * 32, r16, half);
            acc = __builtin_amdgcn_wmma_f32_16x16x32_bf16(false, a1.v, false,
                                                          bfr[kk + 1].v, (short)0, acc,
                                                          false, false);
        }
        {
            const int n = n0 + r16;
#pragma unroll
            for (int r = 0; r < 8; ++r) {
                const int m = half * 8 + r;
                gh[(size_t)m * G3 + n] = acc[r] + bhh[n];
            }
        }
        grid_barrier(bar, GRU_BLOCKS);

        // ---- phase 2: gate fusion over 16*1024 elements ----
        int c2 = 0;
        for (int idx = gtid; idx < BB * HH; idx += GRU_NT, ++c2) {
            const int b = idx >> 10, j = idx & 1023;
            const size_t gx = ((size_t)(b * TT + t)) * G3;
            const float xr = gatesX[gx + j];
            const float xz = gatesX[gx + HH + j];
            const float xn = gatesX[gx + 2 * HH + j];
            const size_t go = (size_t)b * G3;
            const float hr = gh[go + j];
            const float hz = gh[go + HH + j];
            const float hn = gh[go + 2 * HH + j];
            const float hp = hreg[c2];
            const float rr = 1.0f / (1.0f + __expf(-(xr + hr)));
            const float zz = 1.0f / (1.0f + __expf(-(xz + hz)));
            const float nn = tanhf(xn + rr * hn);
            const float hN = (1.0f - zz) * nn + zz * hp;
            hreg[c2] = hN;
            hB[idx] = f2bf(hN);
            const size_t ho = ((size_t)(b * TT + t)) * HH + j;
            hsF[ho] = hN;
            hsB[ho] = f2bf(hN);
        }
        grid_barrier(bar, GRU_BLOCKS);
    }
}

// ---------------- host-side orchestration ----------------
extern "C" void kernel_launch(void* const* d_in, const int* in_sizes, int n_in,
                              void* d_out, int out_size, void* d_ws, size_t ws_size,
                              hipStream_t stream) {
    const float* x        = (const float*)d_in[0];
    const float* conv_w3  = (const float*)d_in[1];
    const float* conv_b3  = (const float*)d_in[2];
    const float* conv_w5  = (const float*)d_in[3];
    const float* conv_b5  = (const float*)d_in[4];
    const float* conv_w7  = (const float*)d_in[5];
    const float* conv_b7  = (const float*)d_in[6];
    const float* gru_w_ih = (const float*)d_in[7];
    const float* gru_w_hh = (const float*)d_in[8];
    const float* gru_b_ih = (const float*)d_in[9];
    const float* gru_b_hh = (const float*)d_in[10];
    // d_in[11..14] = attention MLP: dead code (softmax over singleton axis == 1.0)
    const float* fi_w1    = (const float*)d_in[15];
    const float* fi_b1    = (const float*)d_in[16];
    const float* fi_w2    = (const float*)d_in[17];
    const float* fi_b2    = (const float*)d_in[18];
    const float* lateral  = (const float*)d_in[19];

    char* ws = (char*)d_ws;
    size_t off = 0;
    auto alloc = [&](size_t bytes) { size_t r = off; off += (bytes + 255) & ~(size_t)255; return r; };

    bh*    featB  = (bh*)   (ws + alloc((size_t)BT * GIN * 2));
    bh*    WihB   = (bh*)   (ws + alloc((size_t)G3 * GIN * 2));
    bh*    WhhB   = (bh*)   (ws + alloc((size_t)G3 * HH * 2));
    bh*    latB   = (bh*)   (ws + alloc((size_t)HH * HH * 2));
    bh*    fi1B   = (bh*)   (ws + alloc((size_t)HH * HH * 2));
    bh*    fi2B   = (bh*)   (ws + alloc((size_t)HH * HH * 2));
    float* gatesX = (float*)(ws + alloc((size_t)BT * G3 * 4));
    float* hsF    = (float*)(ws + alloc((size_t)BT * HH * 4));
    bh*    hsB    = (bh*)   (ws + alloc((size_t)BT * HH * 2));
    bh*    inhibB = (bh*)   (ws + alloc((size_t)BT * HH * 2));
    bh*    midB   = (bh*)   (ws + alloc((size_t)BT * HH * 2));
    bh*    hB     = (bh*)   (ws + alloc((size_t)BB * HH * 2));
    float* gh     = (float*)(ws + alloc((size_t)BB * G3 * 4));
    int*   bar    = (int*)  (ws + alloc(256));
    (void)ws_size; (void)in_sizes; (void)n_in; (void)out_size;

    // --- init + weight conversions to bf16 ---
    k_zero_bar<<<1, 1, 0, stream>>>(bar);
    k_cvt_bf16<<<(G3 * GIN + 255) / 256, 256, 0, stream>>>(gru_w_ih, WihB, G3 * GIN);
    k_cvt_bf16<<<(G3 * HH  + 255) / 256, 256, 0, stream>>>(gru_w_hh, WhhB, G3 * HH);
    k_cvt_bf16<<<(HH * HH  + 255) / 256, 256, 0, stream>>>(fi_w1, fi1B, HH * HH);
    k_cvt_bf16<<<(HH * HH  + 255) / 256, 256, 0, stream>>>(fi_w2, fi2B, HH * HH);
    k_cvt_tr_bf16<<<(HH * HH + 255) / 256, 256, 0, stream>>>(lateral, latB, HH, HH);

    // --- conv + relu + max pool -> feat (BT, 96) bf16 ---
    k_conv_pool<<<BT, 128, 0, stream>>>(x, conv_w3, conv_b3, conv_w5, conv_b5,
                                        conv_w7, conv_b7, featB);

    // --- gates_x = feat @ W_ih^T + b_ih :  (4096 x 3072), K=96 ---
    k_gemm_bf16<0><<<dim3(G3 / 64, BT / 16), 128, 0, stream>>>(
        featB, GIN, WihB, GIN, gru_b_ih, nullptr, gatesX, nullptr, G3, GIN);

    // --- full GRU recurrence: ONE persistent kernel, internal grid barriers ---
    k_gru_persist<<<GRU_BLOCKS, GRU_THREADS, 0, stream>>>(
        WhhB, gru_b_hh, gatesX, hB, hsF, hsB, gh, bar);

    // --- attended == hs. lateral inhibition: inhib = hs - relu(hs @ lateral) ---
    k_gemm_bf16<1><<<dim3(HH / 64, BT / 16), 128, 0, stream>>>(
        hsB, HH, latB, HH, nullptr, hsF, nullptr, inhibB, HH, HH);

    // --- mid = relu(inhib @ fi_w1^T + fi_b1) ---
    k_gemm_bf16<2><<<dim3(HH / 64, BT / 16), 128, 0, stream>>>(
        inhibB, HH, fi1B, HH, fi_b1, nullptr, nullptr, midB, HH, HH);

    // --- features = mid @ fi_w2^T + fi_b2 -> d_out[0 : BT*H] fp32 ---
    k_gemm_bf16<0><<<dim3(HH / 64, BT / 16), 128, 0, stream>>>(
        midB, HH, fi2B, HH, fi_b2, nullptr, (float*)d_out, nullptr, HH, HH);

    // --- attention weights output: exactly 1.0 (softmax over length-1 axis) ---
    k_ones<<<(BT + 255) / 256, 256, 0, stream>>>((float*)d_out + (size_t)BT * HH, BT);
}
